// MultiheadAttentionQuant_19739669692754
// MI455X (gfx1250) — matrix operand order repaired and checked by reading, not compiled, not run on hardware
//
#include <hip/hip_runtime.h>
#include <hip/hip_bf16.h>
#include <cstdint>
#include <cstddef>

// ---------------- types ----------------
typedef __attribute__((ext_vector_type(16))) __bf16 v16bf;
typedef __attribute__((ext_vector_type(8)))  __bf16 v8bf;
typedef __attribute__((ext_vector_type(8)))  float  v8f;

#define DEV __device__ __forceinline__

// Monotone float <-> uint32 encoding so integer atomicMin/Max give float min/max.
DEV unsigned fenc(float f) { unsigned u = __float_as_uint(f); return (u & 0x80000000u) ? ~u : (u | 0x80000000u); }
DEV float    fdec(unsigned u) { unsigned v = (u & 0x80000000u) ? (u & 0x7fffffffu) : ~u; return __uint_as_float(v); }

// A-operand fragment (16x32 bf16 tile, row-major, stride in elements).
// ISA layout: lane m=lane&15 is matrix row; lanes>=16 use K base +8;
// VGPR0-3 = K[kb..kb+7], VGPR4-7 = K[kb+16..kb+23].
DEV v16bf frag_a(const __bf16* tile, size_t stride, int lane) {
  int m  = lane & 15;
  int kb = (lane >> 4) << 3;
  const __bf16* p = tile + (size_t)m * stride + kb;
  v8bf lo = *(const v8bf*)(p);
  v8bf hi = *(const v8bf*)(p + 16);
  return __builtin_shufflevector(lo, hi, 0,1,2,3,4,5,6,7,8,9,10,11,12,13,14,15);
}

// B-operand fragment (32x16 bf16, sourced from a row-major [N,K] tile).
// ISA layout: lane n=lane&15 is matrix column; lanes>=16 use K base +8;
// VGPR0-3 = K[kb..kb+7], VGPR4-7 = K[kb+16..kb+23].
DEV v16bf frag_b(const __bf16* tile, size_t stride, int lane) {
  int n  = lane & 15;
  int kh = (lane >> 4) << 3;
  const __bf16* p = tile + (size_t)n * stride + kh;
  v8bf lo = *(const v8bf*)(p);
  v8bf hi = *(const v8bf*)(p + 16);
  return __builtin_shufflevector(lo, hi, 0,1,2,3,4,5,6,7,8,9,10,11,12,13,14,15);
}

DEV v8f wmma_bf16(v16bf a, v16bf b, v8f c) {
  return __builtin_amdgcn_wmma_f32_16x16x32_bf16(false, a, false, b, (short)0, c, false, false);
}

DEV float fq(float x, float mn, float mx) {           // fake-quant, k=1 -> s = mx-mn
  float s = mx - mn;
  return truncf(fminf(fmaxf(x, mn), mx) / s) * s + mn;
}

// ---------------- slot init ----------------
// slots[0..7]  per-tensor (qmin,qmax,kmin,kmax,vmin,vmax,xmin,xmax)
// slots[8..]   per-head blocks of 32 (16 min then 16 max): Q@8, K@40, V@72, P@104
__global__ void k_init_slots(unsigned* sl) {
  int t = threadIdx.x;
  if (t < 136) {
    bool isMin = (t < 8) ? ((t & 1) == 0) : ((((t - 8) >> 4) & 1) == 0);
    sl[t] = isMin ? 0xFFFFFFFFu : 0u;
  }
}

// ---------------- per-tensor min/max ----------------
__global__ __launch_bounds__(256) void k_minmax_tensor(const float4* __restrict__ x, int n4, unsigned* __restrict__ sl) {
  float mn = __builtin_inff(), mx = -__builtin_inff();
  for (int i = blockIdx.x * blockDim.x + threadIdx.x; i < n4; i += gridDim.x * blockDim.x) {
    float4 v = x[i];
    mn = fminf(mn, fminf(fminf(v.x, v.y), fminf(v.z, v.w)));
    mx = fmaxf(mx, fmaxf(fmaxf(v.x, v.y), fmaxf(v.z, v.w)));
  }
  #pragma unroll
  for (int m = 16; m; m >>= 1) { mn = fminf(mn, __shfl_xor(mn, m, 32)); mx = fmaxf(mx, __shfl_xor(mx, m, 32)); }
  if ((threadIdx.x & 31) == 0) { atomicMin(&sl[0], fenc(mn)); atomicMax(&sl[1], fenc(mx)); }
}

// ---------------- per-head min/max over [8192,1024] fp32 ----------------
__global__ __launch_bounds__(256) void k_minmax_head(const float* __restrict__ Y, unsigned* __restrict__ sl) {
  const int D = 1024, R = 8192;
  int col = (blockIdx.x & 3) * 256 + threadIdx.x;
  int rstride = gridDim.x >> 2;
  float mn = __builtin_inff(), mx = -__builtin_inff();
  for (int r = (blockIdx.x >> 2); r < R; r += rstride) {
    float v = Y[(size_t)r * D + col];
    mn = fminf(mn, v); mx = fmaxf(mx, v);
  }
  #pragma unroll
  for (int m = 16; m; m >>= 1) { mn = fminf(mn, __shfl_xor(mn, m, 32)); mx = fmaxf(mx, __shfl_xor(mx, m, 32)); }
  if ((threadIdx.x & 31) == 0) { int h = col >> 6; atomicMin(&sl[h], fenc(mn)); atomicMax(&sl[16 + h], fenc(mx)); }
}

// ---------------- elementwise quant / convert ----------------
__global__ __launch_bounds__(256) void k_quant_t(const float* __restrict__ x, const unsigned* __restrict__ sl,
                                                 __bf16* __restrict__ out, int n) {
  float mn = fdec(sl[0]), mx = fdec(sl[1]);
  for (int i = blockIdx.x * blockDim.x + threadIdx.x; i < n; i += gridDim.x * blockDim.x)
    out[i] = (__bf16)fq(x[i], mn, mx);
}

__global__ __launch_bounds__(256) void k_quant_head(const float* __restrict__ Y, const unsigned* __restrict__ sl,
                                                    __bf16* __restrict__ out, int n) {
  for (int i = blockIdx.x * blockDim.x + threadIdx.x; i < n; i += gridDim.x * blockDim.x) {
    int h = (i & 1023) >> 6;
    out[i] = (__bf16)fq(Y[i], fdec(sl[h]), fdec(sl[16 + h]));
  }
}

// quant V per-head and write transposed [B,H,64,S] so PV B-frags load contiguous.
__global__ __launch_bounds__(256) void k_quant_headT(const float* __restrict__ Y, const unsigned* __restrict__ sl,
                                                     __bf16* __restrict__ out, int n) {
  for (int i = blockIdx.x * blockDim.x + threadIdx.x; i < n; i += gridDim.x * blockDim.x) {
    int col = i & 1023, row = i >> 10;
    int h = col >> 6, d = col & 63, b = row >> 11, s = row & 2047;
    float q = fq(Y[i], fdec(sl[h]), fdec(sl[16 + h]));
    out[(((size_t)(b * 16 + h) << 6) + d) * 2048 + s] = (__bf16)q;
  }
}

__global__ __launch_bounds__(256) void k_f32_bf16(const float* __restrict__ x, __bf16* __restrict__ out, int n) {
  for (int i = blockIdx.x * blockDim.x + threadIdx.x; i < n; i += gridDim.x * blockDim.x)
    out[i] = (__bf16)x[i];
}

// ---------------- GEMM: C[M,N] = A[M,K](bf16) * B[N,K]^T(bf16) + bias, fp32 out ----------------
// 8 waves/block as 4(M)x2(N); wave tile 64x64 = 4x4 WMMA tiles; K step 32.
__global__ __launch_bounds__(256) void k_gemm_bias(const __bf16* __restrict__ A, const __bf16* __restrict__ Bm,
                                                   const float* __restrict__ bias, float* __restrict__ C,
                                                   int M, int N, int K) {
  int lane = threadIdx.x & 31, wave = threadIdx.x >> 5;
  int m0 = blockIdx.y * 256 + (wave & 3) * 64;
  int n0 = blockIdx.x * 128 + (wave >> 2) * 64;
  v8f acc[4][4];
  #pragma unroll
  for (int i = 0; i < 4; i++)
    #pragma unroll
    for (int j = 0; j < 4; j++) acc[i][j] = (v8f){0.f,0.f,0.f,0.f,0.f,0.f,0.f,0.f};

  for (int k = 0; k < K; k += 32) {
    v16bf af[4], bf[4];
    #pragma unroll
    for (int t = 0; t < 4; t++) af[t] = frag_a(A + (size_t)(m0 + t * 16) * K + k, K, lane);
    #pragma unroll
    for (int t = 0; t < 4; t++) bf[t] = frag_b(Bm + (size_t)(n0 + t * 16) * K + k, K, lane);
    #pragma unroll
    for (int mt = 0; mt < 4; mt++)
      #pragma unroll
      for (int nt = 0; nt < 4; nt++) acc[mt][nt] = wmma_bf16(af[mt], bf[nt], acc[mt][nt]);
  }
  int nn = lane & 15, mo = (lane >> 4) << 3;
  #pragma unroll
  for (int mt = 0; mt < 4; mt++)
    #pragma unroll
    for (int nt = 0; nt < 4; nt++) {
      int col = n0 + nt * 16 + nn;
      float bv = bias[col];
      #pragma unroll
      for (int r = 0; r < 8; r++)
        C[(size_t)(m0 + mt * 16 + mo + r) * N + col] = acc[mt][nt][r] + bv;
    }
}

// ---------------- attention pass 1: row max / sum-exp / min (flash-style) ----------------
__global__ __launch_bounds__(256) void k_attn_stats(const __bf16* __restrict__ Qb, const __bf16* __restrict__ Kb,
                                                    float* __restrict__ rowM, float* __restrict__ rowL,
                                                    float* __restrict__ rowSm, unsigned* __restrict__ pSl) {
  const int S = 2048, D = 1024;
  int lane = threadIdx.x & 31;
  int w = blockIdx.x * 8 + (threadIdx.x >> 5);
  int tile = w & 127, bh = w >> 7, b = bh >> 4, h = bh & 15, s0 = tile << 4;

  const __bf16* qt = Qb + ((size_t)b * S + s0) * D + h * 64;
  v16bf a0 = frag_a(qt, D, lane);
  v16bf a1 = frag_a(qt + 32, D, lane);

  float m[8], l[8], mn[8];
  #pragma unroll
  for (int r = 0; r < 8; r++) { m[r] = -__builtin_inff(); l[r] = 0.f; mn[r] = __builtin_inff(); }

  for (int ct = 0; ct < 128; ct++) {
    const __bf16* kt = Kb + ((size_t)b * S + ct * 16) * D + h * 64;
    v8f c = (v8f){0.f,0.f,0.f,0.f,0.f,0.f,0.f,0.f};
    c = wmma_bf16(a0, frag_b(kt, D, lane), c);
    c = wmma_bf16(a1, frag_b(kt + 32, D, lane), c);
    #pragma unroll
    for (int r = 0; r < 8; r++) {
      float s = c[r] * 0.125f;                       // / sqrt(dk)
      mn[r] = fminf(mn[r], s);
      float mx = fmaxf(m[r], s);
      l[r] = l[r] * __expf(m[r] - mx) + __expf(s - mx);
      m[r] = mx;
    }
  }
  // merge across the 16 lanes holding different columns of the same rows
  #pragma unroll
  for (int msk = 8; msk; msk >>= 1)
    #pragma unroll
    for (int r = 0; r < 8; r++) {
      float om = __shfl_xor(m[r], msk, 16), ol = __shfl_xor(l[r], msk, 16), os = __shfl_xor(mn[r], msk, 16);
      float mx = fmaxf(m[r], om);
      l[r] = l[r] * __expf(m[r] - mx) + ol * __expf(om - mx);
      m[r] = mx; mn[r] = fminf(mn[r], os);
    }
  if ((lane & 15) == 0) {
    size_t base = (size_t)bh * S + s0 + ((lane >> 4) << 3);
    #pragma unroll
    for (int r = 0; r < 8; r++) {
      rowM[base + r] = m[r]; rowL[base + r] = l[r]; rowSm[base + r] = mn[r];
      float pmx = 1.f / l[r];
      float pmn = __expf(mn[r] - m[r]) / l[r];
      atomicMin(&pSl[h], fenc(pmn));
      atomicMax(&pSl[16 + h], fenc(pmx));
    }
  }
}

// ---------------- attention pass 2: p -> quant -> (LDS transpose) -> P@V ----------------
__global__ __launch_bounds__(256) void k_attn_out(const __bf16* __restrict__ Qb, const __bf16* __restrict__ Kb,
                                                  const __bf16* __restrict__ Vt, const float* __restrict__ rowM,
                                                  const float* __restrict__ rowL, const unsigned* __restrict__ pSl,
                                                  float* __restrict__ X) {
  const int S = 2048, D = 1024;
  __shared__ __bf16 lds[8 * 512];                    // per-wave 16x32 bf16 staging
  int lane = threadIdx.x & 31, wave = threadIdx.x >> 5;
  int w = blockIdx.x * 8 + wave;
  int tile = w & 127, bh = w >> 7, b = bh >> 4, h = bh & 15, s0 = tile << 4;
  __bf16* P = &lds[wave * 512];

  const __bf16* qt = Qb + ((size_t)b * S + s0) * D + h * 64;
  v16bf a0 = frag_a(qt, D, lane);
  v16bf a1 = frag_a(qt + 32, D, lane);

  int iOff = (lane >> 4) << 3, jc = lane & 15;
  size_t rb = (size_t)bh * S + s0 + iOff;
  float m8[8], li8[8];
  #pragma unroll
  for (int r = 0; r < 8; r++) { m8[r] = rowM[rb + r]; li8[r] = 1.f / rowL[rb + r]; }
  float pmn = fdec(pSl[h]), pmx = fdec(pSl[16 + h]);
  float sq = pmx - pmn;

  v8f xacc[4];
  #pragma unroll
  for (int t = 0; t < 4; t++) xacc[t] = (v8f){0.f,0.f,0.f,0.f,0.f,0.f,0.f,0.f};

  const __bf16* vbase = Vt + ((size_t)bh << 6) * S;
  for (int ct = 0; ct < 128; ct++) {
    const __bf16* kt = Kb + ((size_t)b * S + ct * 16) * D + h * 64;
    v8f c = (v8f){0.f,0.f,0.f,0.f,0.f,0.f,0.f,0.f};
    c = wmma_bf16(a0, frag_b(kt, D, lane), c);
    c = wmma_bf16(a1, frag_b(kt + 32, D, lane), c);
    #pragma unroll
    for (int r = 0; r < 8; r++) {
      float s = c[r] * 0.125f;
      float p = __expf(s - m8[r]) * li8[r];
      float pc = fminf(fmaxf(p, pmn), pmx);
      float q = truncf(pc / sq) * sq + pmn;
      P[(iOff + r) * 32 + ((ct & 1) << 4) + jc] = (__bf16)q;
    }
    if (ct & 1) {                                    // uniform branch: EXEC stays all-ones
      asm volatile("s_wait_dscnt 0" ::: "memory");
      v16bf pa = frag_a(P, 32, lane);                // transpose via A-layout DS reads
      int sc = (ct >> 1) << 5;
      #pragma unroll
      for (int nt = 0; nt < 4; nt++)
        xacc[nt] = wmma_bf16(pa, frag_b(vbase + (size_t)(nt * 16) * S + sc, S, lane), xacc[nt]);
    }
  }
  #pragma unroll
  for (int nt = 0; nt < 4; nt++)
    #pragma unroll
    for (int r = 0; r < 8; r++)
      X[((size_t)b * S + s0 + iOff + r) * D + h * 64 + nt * 16 + jc] = xacc[nt][r];
}

// ---------------- host ----------------
extern "C" void kernel_launch(void* const* d_in, const int* in_sizes, int n_in,
                              void* d_out, int out_size, void* d_ws, size_t ws_size,
                              hipStream_t stream) {
  (void)in_sizes; (void)n_in; (void)out_size; (void)ws_size;
  const float* q  = (const float*)d_in[0];
  const float* ky = (const float*)d_in[1];
  const float* vl = (const float*)d_in[2];
  const float* Wq = (const float*)d_in[3];
  const float* bq = (const float*)d_in[4];
  const float* Wk = (const float*)d_in[5];
  const float* bk = (const float*)d_in[6];
  const float* Wv = (const float*)d_in[7];
  const float* bv = (const float*)d_in[8];
  const float* Wo = (const float*)d_in[9];
  const float* bo = (const float*)d_in[10];
  float* out = (float*)d_out;
  char* ws = (char*)d_ws;

  const int Btot = 4, S = 2048, D = 1024, M = Btot * S;
  const size_t NELT = (size_t)M * D;                // 8388608

  unsigned* slots = (unsigned*)ws;                  // 136 u32 min/max slots
  float* rowM  = (float*)(ws + (4u << 10));
  float* rowL  = rowM + (size_t)64 * S;
  float* rowSm = rowL + (size_t)64 * S;
  __bf16* wqB = (__bf16*)(ws + (2u << 20));
  __bf16* wkB = wqB + (size_t)D * D;
  __bf16* wvB = wkB + (size_t)D * D;
  __bf16* woB = wvB + (size_t)D * D;
  __bf16* actQ = (__bf16*)(ws + (10u << 20));       // 16MB each
  __bf16* actK = actQ + NELT;
  __bf16* actV = actK + NELT;
  float* Yq = (float*)(ws + (58u << 20));           // 32MB each
  float* Yk = Yq + NELT;
  float* Yv = Yk + NELT;
  // buffer reuse (stream-ordered, safe): projections consume act*, then:
  __bf16* Qb = actQ; __bf16* Kb = actK; __bf16* Vt = actV;
  float* X = Yq;                                    // attention output (Yq free by then)
  __bf16* xq = (__bf16*)Yk;                         // quantized x (Yk free by then)

  k_init_slots<<<1, 256, 0, stream>>>(slots);

  int n4 = (int)(NELT / 4);
  k_minmax_tensor<<<512, 256, 0, stream>>>((const float4*)q,  n4, slots + 0);
  k_minmax_tensor<<<512, 256, 0, stream>>>((const float4*)ky, n4, slots + 2);
  k_minmax_tensor<<<512, 256, 0, stream>>>((const float4*)vl, n4, slots + 4);
  k_quant_t<<<2048, 256, 0, stream>>>(q,  slots + 0, actQ, (int)NELT);
  k_quant_t<<<2048, 256, 0, stream>>>(ky, slots + 2, actK, (int)NELT);
  k_quant_t<<<2048, 256, 0, stream>>>(vl, slots + 4, actV, (int)NELT);

  k_f32_bf16<<<1024, 256, 0, stream>>>(Wq, wqB, D * D);
  k_f32_bf16<<<1024, 256, 0, stream>>>(Wk, wkB, D * D);
  k_f32_bf16<<<1024, 256, 0, stream>>>(Wv, wvB, D * D);
  k_f32_bf16<<<1024, 256, 0, stream>>>(Wo, woB, D * D);

  dim3 gg(D / 128, M / 256);                        // (8, 32)
  k_gemm_bias<<<gg, 256, 0, stream>>>(actQ, wqB, bq, Yq, M, D, D);
  k_gemm_bias<<<gg, 256, 0, stream>>>(actK, wkB, bk, Yk, M, D, D);
  k_gemm_bias<<<gg, 256, 0, stream>>>(actV, wvB, bv, Yv, M, D, D);

  k_minmax_head<<<1024, 256, 0, stream>>>(Yq, slots + 8);
  k_minmax_head<<<1024, 256, 0, stream>>>(Yk, slots + 40);
  k_minmax_head<<<1024, 256, 0, stream>>>(Yv, slots + 72);
  k_quant_head<<<2048, 256, 0, stream>>>(Yq, slots + 8,  Qb, (int)NELT);
  k_quant_head<<<2048, 256, 0, stream>>>(Yk, slots + 40, Kb, (int)NELT);
  k_quant_headT<<<2048, 256, 0, stream>>>(Yv, slots + 72, Vt, (int)NELT);

  k_attn_stats<<<1024, 256, 0, stream>>>(Qb, Kb, rowM, rowL, rowSm, slots + 104);
  k_attn_out<<<1024, 256, 0, stream>>>(Qb, Kb, Vt, rowM, rowL, slots + 104, X);

  k_minmax_tensor<<<512, 256, 0, stream>>>((const float4*)X, n4, slots + 6);
  k_quant_t<<<2048, 256, 0, stream>>>(X, slots + 6, xq, (int)NELT);
  k_gemm_bias<<<gg, 256, 0, stream>>>(xq, woB, bo, out, M, D, D);
}
